// Flow_Attention_Causal_3427383902647
// MI455X (gfx1250) — compile-verified, matrix operand order checked by reading, not run on hardware
//
#include <hip/hip_runtime.h>
#include <hip/hip_bf16.h>

typedef __attribute__((ext_vector_type(16))) _Float16 v16h;
typedef __attribute__((ext_vector_type(8)))  _Float16 v8h;
typedef __attribute__((ext_vector_type(4)))  _Float16 v4h;
typedef __attribute__((ext_vector_type(8)))  float    v8f;

#define FA_EPS   1e-6f
#define DH       64
#define CH       64
#define NCH      16
#define LSEQ     1024
#define HEADS    8
#define DMODEL   512
#define BATCH    2
#define BH       (BATCH*HEADS)

static __device__ inline v8f wmma16(v16h a, v16h b, v8f c) {
  return __builtin_amdgcn_wmma_f32_16x16x32_f16(false, a, false, b, (short)0, c,
                                                false, false);
}
static __device__ inline float sigmoidf_(float x) { return 1.0f / (1.0f + __expf(-x)); }

// A/B fragment from f16 row-major storage (global or LDS). ISA 7.12.2 16-bit
// layout: per lane (row = lane%16, half = lane/16) halves K = half*8 + [0..7]
// and K = 16 + half*8 + [0..7] -> two aligned 128-bit loads.
static __device__ inline v16h frag_row16(const _Float16* src, int ld, int base_i, int base_k) {
  int lane = threadIdx.x & 31;
  const _Float16* p = src + (size_t)(base_i + (lane & 15)) * ld + base_k + (lane >> 4) * 8;
  v8h lo = *(const v8h*)p;
  v8h hi = *(const v8h*)(p + 16);
  v16h f;
#pragma unroll
  for (int i = 0; i < 8; ++i) { f[i] = lo[i]; f[8 + i] = hi[i]; }
  return f;
}

// CDNA5 async global->LDS copy (ASYNCcnt) + wait. LDS byte address = low 32
// bits of the generic pointer.
static __device__ inline void async_copy16(const _Float16* gsrc, _Float16* ldst) {
  unsigned laddr = (unsigned)(unsigned long long)ldst;
  asm volatile("global_load_async_to_lds_b128 %0, %1, off"
               :: "v"(laddr), "v"(gsrc) : "memory");
}
static __device__ inline void wait_async0() {
  asm volatile("s_wait_asynccnt 0x0" ::: "memory");
}

// ---- f32 -> f16 vectorized convert -----------------------------------------
__global__ void cvt16_kernel(const float* __restrict__ in, _Float16* __restrict__ out,
                             int n4) {
  int i = blockIdx.x * blockDim.x + threadIdx.x;
  if (i >= n4) return;
  float4 x = ((const float4*)in)[i];
  v4h o;
  o[0] = (_Float16)x.x; o[1] = (_Float16)x.y; o[2] = (_Float16)x.z; o[3] = (_Float16)x.w;
  ((v4h*)out)[i] = o;
}

// ---- weight transpose+convert: out[n][k] = (f16)in[k][n], 512x512 ----------
__global__ void wtrans_kernel(const float* __restrict__ in, _Float16* __restrict__ out) {
  __shared__ float tile[32][33];
  int tx = threadIdx.x & 31, ty = threadIdx.x >> 5;   // 256 thr = 32x8
  int r0 = blockIdx.y * 32, c0 = blockIdx.x * 32;
  for (int i = ty; i < 32; i += 8)
    tile[i][tx] = in[(size_t)(r0 + i) * DMODEL + c0 + tx];
  __syncthreads();
  for (int i = ty; i < 32; i += 8)
    out[(size_t)(c0 + i) * DMODEL + r0 + tx] = (_Float16)tile[tx][i];
}

// ---- per-head transpose (+optional row scale): out[bh][d][l] ---------------
__global__ void head_transpose_kernel(const float* __restrict__ X,
                                      const float* __restrict__ scale,
                                      _Float16* __restrict__ out, int use_scale) {
  int bh = blockIdx.x, h = bh & 7, b = bh >> 3;
  int d = threadIdx.x & 63, g = threadIdx.x >> 6;     // 256 thr
  for (int l = g; l < LSEQ; l += 4) {
    float x = X[((size_t)(b * LSEQ + l)) * DMODEL + h * DH + d];
    if (use_scale) x *= scale[(size_t)bh * LSEQ + l];
    out[((size_t)bh * DH + d) * LSEQ + l] = (_Float16)x;
  }
}

// ---- GEMM: Y = act(X16 * WT16^T + b); async double-buffered LDS staging ----
// block 128 (4 waves); WG tile 64(M) x 64(N); k-step 32.
__global__ void proj_kernel(const _Float16* __restrict__ X16,
                            const _Float16* __restrict__ WT16,
                            const float* __restrict__ bias,
                            float* __restrict__ Yf, _Float16* __restrict__ Yh,
                            int M, int K, int N, int apply_sigmoid,
                            int write_f, int write_h) {
  __shared__ __align__(16) _Float16 Xs[2][64 * 32];
  __shared__ __align__(16) _Float16 Ws[2][64 * 32];
  int tid = threadIdx.x, wv = tid >> 5, lane = tid & 31;
  int row0 = blockIdx.x * 64;
  int col0 = blockIdx.y * 64;

  // Stage one 64x32-half tile of X and of W^T into LDS buffer `buf`.
  // 256 16-byte chunks per tile; 128 threads x 2 chunks; all via async DMA.
  auto stage = [&](int buf, int kk) {
#pragma unroll
    for (int s = 0; s < 2; ++s) {
      int chunk = tid + s * 128;          // 0..255
      int rrow = chunk >> 2, part = chunk & 3;
      async_copy16(X16 + (size_t)(row0 + rrow) * K + kk + part * 8,
                   &Xs[buf][rrow * 32 + part * 8]);
      async_copy16(WT16 + (size_t)(col0 + rrow) * K + kk + part * 8,
                   &Ws[buf][rrow * 32 + part * 8]);
    }
  };

  stage(0, 0);
  wait_async0();
  __syncthreads();

  v8f acc[4] = {};
  int buf = 0;
  for (int kk = 0; kk < K; kk += 32) {
    if (kk + 32 < K) stage(buf ^ 1, kk + 32);   // overlap DMA with compute
    v16h a = frag_row16(Xs[buf], 32, wv * 16, 0);
#pragma unroll
    for (int t = 0; t < 4; ++t) {
      v16h b = frag_row16(Ws[buf], 32, 16 * t, 0);
      acc[t] = wmma16(a, b, acc[t]);
    }
    wait_async0();
    __syncthreads();
    buf ^= 1;
  }

  int n_lo = lane & 15, m_hi = (lane >> 4) * 8;
#pragma unroll
  for (int t = 0; t < 4; ++t) {
    int n = col0 + 16 * t + n_lo;
    float bv = bias[n];
#pragma unroll
    for (int r = 0; r < 8; ++r) {
      int m = row0 + wv * 16 + m_hi + r;
      float y = acc[t][r] + bv;
      if (apply_sigmoid) y = sigmoidf_(y);
      if (write_f) Yf[(size_t)m * N + n] = y;
      if (write_h) Yh[(size_t)m * N + n] = (_Float16)y;
    }
  }
}

// ---- per-(b,h,l) row sums over head dims -----------------------------------
__global__ void row_sum_kernel(const float* __restrict__ X, float* __restrict__ out) {
  int bh = blockIdx.x, h = bh & 7, b = bh >> 3;
  for (int l = threadIdx.x; l < LSEQ; l += blockDim.x) {
    const float* p = X + ((size_t)(b * LSEQ + l)) * DMODEL + h * DH;
    float s = 0.f;
    for (int d = 0; d < DH; ++d) s += p[d];
    out[(size_t)bh * LSEQ + l] = s;
  }
}

// ---- inclusive cumsum of (optionally weighted) length-1024 sequences -------
__global__ void cumsum1024_kernel(const float* __restrict__ in, const float* __restrict__ w,
                                  float* __restrict__ out, int use_w) {
  __shared__ float tot[256];
  int bh = blockIdx.x, t = threadIdx.x, base = t * 4;
  const float* pin = in + (size_t)bh * LSEQ;
  const float* pw  = w  + (size_t)bh * LSEQ;
  float vloc[4]; float s = 0.f;
#pragma unroll
  for (int i = 0; i < 4; ++i) {
    float x = pin[base + i];
    if (use_w) x *= pw[base + i];
    s += x; vloc[i] = s;
  }
  tot[t] = s;
  __syncthreads();
  for (int off = 1; off < 256; off <<= 1) {
    float add = (t >= off) ? tot[t - off] : 0.f;
    __syncthreads();
    tot[t] += add;
    __syncthreads();
  }
  float excl = tot[t] - s;
  float* pout = out + (size_t)bh * LSEQ;
#pragma unroll
  for (int i = 0; i < 4; ++i) pout[base + i] = vloc[i] + excl;
}

// ---- per-chunk vector sums (optionally row-weighted) -----------------------
__global__ void chunk_vec_sum_kernel(const float* __restrict__ X, const float* __restrict__ w,
                                     float* __restrict__ out, int use_w) {
  int idx = blockIdx.x, c = idx & 15, bh = idx >> 4, h = bh & 7, b = bh >> 3;
  int d = threadIdx.x;
  const float* base = X + ((size_t)(b * LSEQ + c * CH)) * DMODEL + h * DH + d;
  const float* wb = w + (size_t)bh * LSEQ + c * CH;
  float s = 0.f;
  for (int r = 0; r < CH; ++r) {
    float xv = base[(size_t)r * DMODEL];
    s += use_w ? xv * wb[r] : xv;
  }
  out[(size_t)idx * DH + d] = s;
}

// ---- exclusive prefix over the 16 chunks (in place) ------------------------
__global__ void chunk_exprefix_kernel(float* data, int elems) {
  int bh = blockIdx.x;
  for (int e = threadIdx.x; e < elems; e += blockDim.x) {
    float run = 0.f;
    for (int c2 = 0; c2 < NCH; ++c2) {
      size_t i = ((size_t)bh * NCH + c2) * elems + e;
      float vv = data[i];
      data[i] = run;
      run += vv;
    }
  }
}

// ---- pass 1: S = Q_c K_c^T (WMMA), sink_incoming / source_outgoing ---------
__global__ void pass1_kernel(const float* __restrict__ q, const float* __restrict__ k,
                             const _Float16* __restrict__ q16, const _Float16* __restrict__ k16,
                             const float* __restrict__ kpre, const float* __restrict__ qpre,
                             const float* __restrict__ rsq, const float* __restrict__ rsk,
                             const float* __restrict__ crk, const float* __restrict__ crq,
                             float* __restrict__ Sbuf, float* __restrict__ si,
                             float* __restrict__ so) {
  __shared__ float Sl[CH][CH + 1];
  __shared__ float kpre_s[DH], qpre_s[DH];
  int idx = blockIdx.x, c = idx & 15, bh = idx >> 4, h = bh & 7, b = bh >> 3;
  size_t rowbase = (size_t)(b * LSEQ + c * CH);
  const float*    qc  = q   + rowbase * DMODEL + h * DH;
  const float*    kc  = k   + rowbase * DMODEL + h * DH;
  const _Float16* qch = q16 + rowbase * DMODEL + h * DH;
  const _Float16* kch = k16 + rowbase * DMODEL + h * DH;
  int wv = threadIdx.x >> 5, lane = threadIdx.x & 31;
  if (threadIdx.x < DH) {
    kpre_s[threadIdx.x] = kpre[(size_t)idx * DH + threadIdx.x];
    qpre_s[threadIdx.x] = qpre[(size_t)idx * DH + threadIdx.x];
  }
  v8f acc[4] = {};
  for (int dk = 0; dk < DH; dk += 32) {
    v16h a = frag_row16(qch, DMODEL, wv * 16, dk);
#pragma unroll
    for (int t = 0; t < 4; ++t) {
      v16h bf = frag_row16(kch, DMODEL, t * 16, dk);   // B = K^T
      acc[t] = wmma16(a, bf, acc[t]);
    }
  }
  int n_lo = lane & 15, m_hi = (lane >> 4) * 8;
  float* Sg = Sbuf + (size_t)idx * (CH * CH);
#pragma unroll
  for (int t = 0; t < 4; ++t)
#pragma unroll
    for (int r = 0; r < 8; ++r) {
      int m = wv * 16 + m_hi + r, n = t * 16 + n_lo;
      Sl[m][n] = acc[t][r];
      Sg[m * CH + n] = acc[t][r];
    }
  __syncthreads();
  int l = threadIdx.x & 63, lg = c * CH + l;
  float normal = (float)(lg + 1);
  size_t soff = (size_t)bh * LSEQ + lg;
  const float e2 = (float)DH * FA_EPS * FA_EPS;
  if (threadIdx.x < 64) {
    const float* qrow = qc + (size_t)l * DMODEL;
    float dq = 0.f;
    for (int d = 0; d < DH; ++d) dq += qrow[d] * kpre_s[d];
    float tr = 0.f;
    for (int j = 0; j <= l; ++j) tr += Sl[l][j];
    si[soff] = normal / (dq + tr + FA_EPS * (rsq[soff] + crk[soff]) + e2);
  } else {
    const float* krow = kc + (size_t)l * DMODEL;
    float dk2 = 0.f;
    for (int d = 0; d < DH; ++d) dk2 += krow[d] * qpre_s[d];
    float tc = 0.f;
    for (int j = 0; j <= l; ++j) tc += Sl[j][l];
    so[soff] = normal / (dk2 + tc + FA_EPS * (rsk[soff] + crq[soff]) + e2);
  }
}

// ---- pass 2: conserved sink/source (reuse stored S) ------------------------
__global__ void pass2_kernel(const float* __restrict__ q, const float* __restrict__ k,
                             const float* __restrict__ ksopre, const float* __restrict__ qsipre,
                             const float* __restrict__ rsq, const float* __restrict__ rsk,
                             const float* __restrict__ crkso, const float* __restrict__ crqsi,
                             const float* __restrict__ si, const float* __restrict__ so,
                             const float* __restrict__ Sbuf, float* __restrict__ salloc,
                             float* __restrict__ expcs) {
  __shared__ float Sl[CH][CH + 1];
  __shared__ float sop_s[CH], sip_s[CH], kso_s[DH], qsi_s[DH];
  int idx = blockIdx.x, c = idx & 15, bh = idx >> 4, h = bh & 7, b = bh >> 3;
  const float* qc = q + ((size_t)(b * LSEQ + c * CH)) * DMODEL + h * DH;
  const float* kc = k + ((size_t)(b * LSEQ + c * CH)) * DMODEL + h * DH;
  const float* Sg = Sbuf + (size_t)idx * (CH * CH);
  for (int e = threadIdx.x; e < CH * CH; e += blockDim.x) Sl[e >> 6][e & 63] = Sg[e];
  if (threadIdx.x < CH) {
    size_t soff = (size_t)bh * LSEQ + c * CH + threadIdx.x;
    sop_s[threadIdx.x] = so[soff];
    sip_s[threadIdx.x] = si[soff];
    kso_s[threadIdx.x] = ksopre[(size_t)idx * DH + threadIdx.x];
    qsi_s[threadIdx.x] = qsipre[(size_t)idx * DH + threadIdx.x];
  }
  __syncthreads();
  int l = threadIdx.x & 63, lg = c * CH + l;
  float normal = (float)(lg + 1);
  size_t soff = (size_t)bh * LSEQ + lg;
  const float e2 = (float)DH * FA_EPS * FA_EPS;
  if (threadIdx.x < 64) {
    const float* qrow = qc + (size_t)l * DMODEL;
    float dq = 0.f;
    for (int d = 0; d < DH; ++d) dq += qrow[d] * kso_s[d];
    float tr = 0.f;
    for (int j = 0; j <= l; ++j) tr += sop_s[j] * Sl[l][j];
    float cs = (dq + tr + FA_EPS * (rsq[soff] + crkso[soff]) + e2) / normal;
    salloc[soff] = sigmoidf_(cs);
  } else {
    const float* krow = kc + (size_t)l * DMODEL;
    float dk2 = 0.f;
    for (int d = 0; d < DH; ++d) dk2 += krow[d] * qsi_s[d];
    float tc = 0.f;
    for (int j = 0; j <= l; ++j) tc += sip_s[j] * Sl[j][l];
    float cv = (dk2 + tc + FA_EPS * (rsk[soff] + crqsi[soff]) + e2) / normal;
    cv = fminf(1.f, fmaxf(-1.f, cv));
    expcs[soff] = __expf(cv);
  }
}

__global__ void scomp_kernel(const float* __restrict__ expcs, const float* __restrict__ ecum,
                             float* __restrict__ sc) {
  size_t i = (size_t)blockIdx.x * blockDim.x + threadIdx.x;
  if (i >= (size_t)BH * LSEQ) return;
  int l = (int)(i & (LSEQ - 1));
  sc[i] = expcs[i] / ecum[i] * (float)(l + 1);
}

// ---- per-chunk KV state: KV_c = K_c^T @ V''_c (WMMA), stored [dv][d] -------
__global__ void kv_chunk_kernel(const _Float16* __restrict__ kT16,
                                const _Float16* __restrict__ vT16s,
                                float* __restrict__ kvbuf) {
  int idx = blockIdx.x, c = idx & 15, bh = idx >> 4;
  const _Float16* kT = kT16  + (size_t)bh * DH * LSEQ;  // [d][l]
  const _Float16* vT = vT16s + (size_t)bh * DH * LSEQ;  // [dv][l], scale folded
  int wv = threadIdx.x >> 5, lane = threadIdx.x & 31;
  v8f acc[4] = {};
  for (int jk = 0; jk < CH; jk += 32) {
    v16h a = frag_row16(kT, LSEQ, wv * 16, c * CH + jk);       // A = K^T (d x j)
#pragma unroll
    for (int t = 0; t < 4; ++t) {
      v16h bf = frag_row16(vT, LSEQ, t * 16, c * CH + jk);     // B = V'' (j x dv)
      acc[t] = wmma16(a, bf, acc[t]);
    }
  }
  float* out = kvbuf + (size_t)idx * (DH * DH);
  int n_lo = lane & 15, m_hi = (lane >> 4) * 8;
#pragma unroll
  for (int t = 0; t < 4; ++t)
#pragma unroll
    for (int r = 0; r < 8; ++r)
      out[(t * 16 + n_lo) * DH + wv * 16 + m_hi + r] = acc[t][r];  // [dv][d]
}

// ---- output pass: tril(diag(si/n)S) @ V'' + Q' @ KVpre, scale, store f16 ---
__global__ void out_pass_kernel(const float* __restrict__ qf,
                                const _Float16* __restrict__ vT16s,
                                const float* __restrict__ Sbuf,
                                const _Float16* __restrict__ kv16,
                                const float* __restrict__ si, const float* __restrict__ salloc,
                                _Float16* __restrict__ att16) {
  __shared__ __align__(16) _Float16 Sm16[CH * CH];
  __shared__ __align__(16) _Float16 Lq16[CH * DH];
  __shared__ float wrow_s[CH], sal_s[CH];
  int idx = blockIdx.x, c = idx & 15, bh = idx >> 4, h = bh & 7, b = bh >> 3;
  const float* qc = qf + ((size_t)(b * LSEQ + c * CH)) * DMODEL + h * DH;
  const float* Sg = Sbuf + (size_t)idx * (CH * CH);
  const _Float16* vT  = vT16s + (size_t)bh * DH * LSEQ;   // [dv][l]
  const _Float16* kvp = kv16 + (size_t)idx * (DH * DH);   // [dv][d]
  if (threadIdx.x < CH) {
    size_t soff = (size_t)bh * LSEQ + c * CH + threadIdx.x;
    wrow_s[threadIdx.x] = si[soff] / (float)(c * CH + threadIdx.x + 1);
    sal_s[threadIdx.x] = salloc[soff];
  }
  __syncthreads();
  for (int e = threadIdx.x; e < CH * CH; e += blockDim.x) {
    int l = e >> 6, j = e & 63;
    Sm16[e] = (_Float16)((j <= l) ? Sg[e] * wrow_s[l] : 0.f);
    Lq16[e] = (_Float16)(qc[(size_t)l * DMODEL + j] * wrow_s[l]);
  }
  __syncthreads();
  int wv = threadIdx.x >> 5, lane = threadIdx.x & 31;
  v8f acc[4] = {};
  for (int jk = 0; jk < CH; jk += 32) {                 // intra-chunk causal part
    v16h a = frag_row16(Sm16, CH, wv * 16, jk);
#pragma unroll
    for (int t = 0; t < 4; ++t) {
      v16h bf = frag_row16(vT, LSEQ, t * 16, c * CH + jk);
      acc[t] = wmma16(a, bf, acc[t]);
    }
  }
  for (int dk = 0; dk < DH; dk += 32) {                 // prefix-state part
    v16h a = frag_row16(Lq16, DH, wv * 16, dk);
#pragma unroll
    for (int t = 0; t < 4; ++t) {
      v16h bf = frag_row16(kvp, DH, t * 16, dk);
      acc[t] = wmma16(a, bf, acc[t]);
    }
  }
  int n_lo = lane & 15, m_hi = (lane >> 4) * 8;
#pragma unroll
  for (int t = 0; t < 4; ++t)
#pragma unroll
    for (int r = 0; r < 8; ++r) {
      int m = wv * 16 + m_hi + r, n = t * 16 + n_lo;
      att16[((size_t)(b * LSEQ + c * CH + m)) * DMODEL + h * DH + n] =
          (_Float16)(acc[t][r] * sal_s[m]);
    }
}

extern "C" void kernel_launch(void* const* d_in, const int* in_sizes, int n_in,
                              void* d_out, int out_size, void* d_ws, size_t ws_size,
                              hipStream_t stream) {
  (void)in_sizes; (void)n_in; (void)out_size; (void)ws_size;
  const float* queries = (const float*)d_in[0];
  const float* keys    = (const float*)d_in[1];
  const float* values  = (const float*)d_in[2];
  const float* Wq = (const float*)d_in[3]; const float* bq = (const float*)d_in[4];
  const float* Wk = (const float*)d_in[5]; const float* bk = (const float*)d_in[6];
  const float* Wv = (const float*)d_in[7]; const float* bv = (const float*)d_in[8];
  const float* Wo = (const float*)d_in[9]; const float* bo = (const float*)d_in[10];
  float* out = (float*)d_out;

  const size_t NQ = (size_t)BATCH * LSEQ * DMODEL;   // 1M elements
  const size_t SB = (size_t)BH * LSEQ;               // 16K elements
  const size_t WSZ = (size_t)DMODEL * DMODEL;        // 256K elements

  float* ws = (float*)d_ws;
  float* q     = ws;          float* k     = q + NQ;
  float* v     = k + NQ;      float* Sbuf  = v + NQ;
  float* kvbuf = Sbuf + NQ;
  float* rsq   = kvbuf + NQ;  float* rsk   = rsq + SB;
  float* crk   = rsk + SB;    float* crq   = crk + SB;
  float* crkso = crq + SB;    float* crqsi = crkso + SB;
  float* si    = crqsi + SB;  float* so    = si + SB;
  float* sal   = so + SB;     float* ecs   = sal + SB;
  float* ecum  = ecs + SB;    float* scm   = ecum + SB;
  float* ksum  = scm + SB;    float* qsum  = ksum + SB;
  float* ksos  = qsum + SB;   float* qsis  = ksos + SB;

  _Float16* hbase = (_Float16*)(qsis + SB);
  _Float16* inq16 = hbase;            _Float16* ink16 = inq16 + NQ;
  _Float16* inv16 = ink16 + NQ;
  _Float16* wtq   = inv16 + NQ;       _Float16* wtk   = wtq + WSZ;
  _Float16* wtv   = wtk + WSZ;        _Float16* wto   = wtv + WSZ;
  _Float16* q16   = wto + WSZ;        _Float16* k16   = q16 + NQ;
  _Float16* att16 = k16 + NQ;
  _Float16* kT16  = att16 + NQ;       _Float16* vT16s = kT16 + NQ;
  _Float16* kv16  = vT16s + NQ;

  const int CV4 = 256;
  cvt16_kernel<<<(int)(NQ / 4 / CV4), CV4, 0, stream>>>(queries, inq16, (int)(NQ / 4));
  cvt16_kernel<<<(int)(NQ / 4 / CV4), CV4, 0, stream>>>(keys,    ink16, (int)(NQ / 4));
  cvt16_kernel<<<(int)(NQ / 4 / CV4), CV4, 0, stream>>>(values,  inv16, (int)(NQ / 4));
  dim3 gt(16, 16);
  wtrans_kernel<<<gt, 256, 0, stream>>>(Wq, wtq);
  wtrans_kernel<<<gt, 256, 0, stream>>>(Wk, wtk);
  wtrans_kernel<<<gt, 256, 0, stream>>>(Wv, wtv);
  wtrans_kernel<<<gt, 256, 0, stream>>>(Wo, wto);

  dim3 gproj(BATCH * LSEQ / 64, DMODEL / 64);
  proj_kernel<<<gproj, 128, 0, stream>>>(inq16, wtq, bq, q, q16,
                                         BATCH * LSEQ, DMODEL, DMODEL, 1, 1, 1);
  proj_kernel<<<gproj, 128, 0, stream>>>(ink16, wtk, bk, k, k16,
                                         BATCH * LSEQ, DMODEL, DMODEL, 1, 1, 1);
  proj_kernel<<<gproj, 128, 0, stream>>>(inv16, wtv, bv, v, att16 /*unused*/,
                                         BATCH * LSEQ, DMODEL, DMODEL, 0, 1, 0);

  row_sum_kernel<<<BH, 256, 0, stream>>>(q, rsq);
  row_sum_kernel<<<BH, 256, 0, stream>>>(k, rsk);
  cumsum1024_kernel<<<BH, 256, 0, stream>>>(rsk, rsk, crk, 0);
  cumsum1024_kernel<<<BH, 256, 0, stream>>>(rsq, rsq, crq, 0);

  chunk_vec_sum_kernel<<<BH * NCH, DH, 0, stream>>>(k, rsk, ksum, 0);
  chunk_vec_sum_kernel<<<BH * NCH, DH, 0, stream>>>(q, rsq, qsum, 0);
  chunk_exprefix_kernel<<<BH, 64, 0, stream>>>(ksum, DH);
  chunk_exprefix_kernel<<<BH, 64, 0, stream>>>(qsum, DH);

  pass1_kernel<<<BH * NCH, 128, 0, stream>>>(q, k, q16, k16, ksum, qsum, rsq, rsk,
                                             crk, crq, Sbuf, si, so);

  chunk_vec_sum_kernel<<<BH * NCH, DH, 0, stream>>>(k, so, ksos, 1);
  chunk_vec_sum_kernel<<<BH * NCH, DH, 0, stream>>>(q, si, qsis, 1);
  chunk_exprefix_kernel<<<BH, 64, 0, stream>>>(ksos, DH);
  chunk_exprefix_kernel<<<BH, 64, 0, stream>>>(qsis, DH);
  cumsum1024_kernel<<<BH, 256, 0, stream>>>(rsk, so, crkso, 1);
  cumsum1024_kernel<<<BH, 256, 0, stream>>>(rsq, si, crqsi, 1);

  pass2_kernel<<<BH * NCH, 128, 0, stream>>>(q, k, ksos, qsis, rsq, rsk, crkso, crqsi,
                                             si, so, Sbuf, sal, ecs);

  cumsum1024_kernel<<<BH, 256, 0, stream>>>(ecs, ecs, ecum, 0);
  scomp_kernel<<<(BH * LSEQ + 255) / 256, 256, 0, stream>>>(ecs, ecum, scm);

  head_transpose_kernel<<<BH, 256, 0, stream>>>(k, scm, kT16, 0);
  head_transpose_kernel<<<BH, 256, 0, stream>>>(v, scm, vT16s, 1);

  kv_chunk_kernel<<<BH * NCH, 128, 0, stream>>>(kT16, vT16s, kvbuf);
  chunk_exprefix_kernel<<<BH, 256, 0, stream>>>(kvbuf, DH * DH);
  cvt16_kernel<<<(int)(NQ / 4 / CV4), CV4, 0, stream>>>(kvbuf, kv16, (int)(NQ / 4));

  out_pass_kernel<<<BH * NCH, 128, 0, stream>>>(q, vT16s, Sbuf, kv16, si, sal, att16);

  proj_kernel<<<gproj, 128, 0, stream>>>(att16, wto, bo, out, att16 /*unused*/,
                                         BATCH * LSEQ, DMODEL, DMODEL, 0, 1, 0);
}
